// SelfAttention_61306363183165
// MI455X (gfx1250) — matrix operand order repaired and checked by reading, not compiled, not run on hardware
//
#include <hip/hip_runtime.h>
#include <math.h>

#define NBATCH 2
#define SEQ 2048
#define EMB 1024
#define NH 16
#define HD 64

typedef __attribute__((ext_vector_type(16))) __bf16 v16bf;
typedef __attribute__((ext_vector_type(8)))  __bf16 v8bf;
typedef __attribute__((ext_vector_type(8)))  float  v8f;

#define WMMA_BF16(A,B,C) __builtin_amdgcn_wmma_f32_16x16x32_bf16(false,(A),false,(B),(short)0,(C),false,false)

static __device__ inline v16bf pack16(v8bf lo, v8bf hi) {
  v16bf r;
#pragma unroll
  for (int j = 0; j < 8; ++j) { r[j] = lo[j]; r[j + 8] = hi[j]; }
  return r;
}

// A fragment (16x32 bf16): lane holds row m=lane&15; element j <-> K = 16*hi + j
// rowp points at this lane's row (already offset by the 32-wide K chunk).
static __device__ inline v16bf load_a_bf16(const __bf16* rowp, int hi) {
  v8bf lo = *(const v8bf*)(rowp + 16 * hi);
  v8bf hh = *(const v8bf*)(rowp + 16 * hi + 8);
  return pack16(lo, hh);
}

// B fragment (32x16 bf16): lane holds col n=lane&15; element j <-> K = 16*hi + j.
// p points at 16 contiguous K values for this lane.
static __device__ inline v16bf load_b_bf16(const __bf16* p) {
  v8bf lo = *(const v8bf*)(p);
  v8bf hh = *(const v8bf*)(p + 8);
  return pack16(lo, hh);
}

static __device__ inline v16bf load_a_f32(const float* rowp, int hi) {
  v16bf r;
#pragma unroll
  for (int j = 0; j < 8; ++j) r[j] = (__bf16)rowp[16 * hi + j];
#pragma unroll
  for (int j = 0; j < 8; ++j) r[j + 8] = (__bf16)rowp[16 * hi + 8 + j];
  return r;
}

static __device__ inline v16bf load_b_f32(const float* p) {
  v16bf r;
#pragma unroll
  for (int j = 0; j < 16; ++j) r[j] = (__bf16)p[j];
  return r;
}

// wave-private LDS read-after-write fence: scheduling barrier + wait on DScnt.
// (LDS ops of one wave complete in order; the fence stops the compiler from
// hoisting the cross-lane ds_loads above the ds_stores.)
static __device__ inline void lds_fence_wave() {
  __builtin_amdgcn_wave_barrier();
  asm volatile("s_wait_dscnt 0x0" ::: "memory");
  __builtin_amdgcn_wave_barrier();
}

// ---------------- Kernel 0: convert Wo to bf16 ----------------
__global__ void k_cvt_wo(const float* __restrict__ wo, __bf16* __restrict__ wob) {
  int i = blockIdx.x * blockDim.x + threadIdx.x;
  wob[i] = (__bf16)wo[i];
}

// ---------------- Kernel 1: shared value projection of V, K, Q ----------------
// out[e] = sum_d x[d] * Wv[e][d] + bv[e], per head.
// One wave = one (source, n, h, 16-row tile). The source select is hoisted to
// the top level so each store loop is branch-free (no per-store EXEC masking).
// Tile loop stays rolled to keep the live set small (no spills).
// Writes qp,kp as [n][h][l][d] bf16 and V transposed vpt as [n][h][d][l] bf16.
__global__ void k_proj(const float* __restrict__ vals,
                       const float* __restrict__ keys,
                       const float* __restrict__ qry,
                       const float* __restrict__ Wv,
                       const float* __restrict__ bv,
                       __bf16* __restrict__ qp,
                       __bf16* __restrict__ kp,
                       __bf16* __restrict__ vpt) {
  const int wave = (blockIdx.x * blockDim.x + threadIdx.x) >> 5;
  const int lane = threadIdx.x & 31;
  const int m = lane & 15;
  const int hi = lane >> 4;

  const int UNITS = NBATCH * NH * (SEQ / 16);  // 4096 per source
  const int src = wave / UNITS;                // 0=values, 1=keys, 2=query
  const int u = wave % UNITS;
  const int lt = u % (SEQ / 16);
  const int h = (u / (SEQ / 16)) % NH;
  const int n = u / ((SEQ / 16) * NH);
  const int l0 = lt * 16;

  const float* x = (src == 0) ? vals : ((src == 1) ? keys : qry);
  const float* base = x + (size_t)(n * SEQ + l0 + m) * EMB + h * HD;
  v16bf a0 = load_a_f32(base, hi);
  v16bf a1 = load_a_f32(base + 32, hi);

  if (src == 0) {
    // V path: store transposed, contiguous in r -> b128 stores
    __bf16* vdst = vpt + (size_t)(n * NH + h) * HD * SEQ;
#pragma unroll 1
    for (int t = 0; t < 4; ++t) {
      v16bf bw0 = load_b_f32(Wv + (size_t)(t * 16 + m) * HD + 0 + hi * 16);
      v16bf bw1 = load_b_f32(Wv + (size_t)(t * 16 + m) * HD + 32 + hi * 16);
      float bias = bv[t * 16 + m];
      v8f c;
#pragma unroll
      for (int r = 0; r < 8; ++r) c[r] = bias;
      c = WMMA_BF16(a0, bw0, c);
      c = WMMA_BF16(a1, bw1, c);
      __bf16* p = vdst + (size_t)(t * 16 + m) * SEQ + l0 + 8 * hi;
#pragma unroll
      for (int r = 0; r < 8; ++r) p[r] = (__bf16)c[r];
    }
  } else {
    // K / Q path: [n][h][l][d] layout (d-contiguous for attention B loads)
    __bf16* dst = ((src == 1) ? kp : qp) + (size_t)(n * NH + h) * SEQ * HD;
#pragma unroll 1
    for (int t = 0; t < 4; ++t) {
      v16bf bw0 = load_b_f32(Wv + (size_t)(t * 16 + m) * HD + 0 + hi * 16);
      v16bf bw1 = load_b_f32(Wv + (size_t)(t * 16 + m) * HD + 32 + hi * 16);
      float bias = bv[t * 16 + m];
      v8f c;
#pragma unroll
      for (int r = 0; r < 8; ++r) c[r] = bias;
      c = WMMA_BF16(a0, bw0, c);
      c = WMMA_BF16(a1, bw1, c);
      __bf16* p = dst + (size_t)(l0 + 8 * hi) * HD + t * 16 + m;
#pragma unroll
      for (int r = 0; r < 8; ++r) p[(size_t)r * HD] = (__bf16)c[r];
    }
  }
}

// ---------------- Kernel 2: flash attention ----------------
// Block = 8 waves; each wave owns a 16-row Q tile (block covers 128 rows) of one (n,h).
// KV streamed in chunks of 32; online softmax; P transposed C->A layout via a
// wave-private LDS slice (no block barrier needed).
__global__ void k_attn(const __bf16* __restrict__ qp,
                       const __bf16* __restrict__ kp,
                       const __bf16* __restrict__ vpt,
                       __bf16* __restrict__ attnb) {
  __shared__ __bf16 plds[8][16 * 32];
  const int w = threadIdx.x >> 5;
  const int lane = threadIdx.x & 31;
  const int m = lane & 15;
  const int hi = lane >> 4;
  const int b = blockIdx.x;
  const int qb = b % (SEQ / 128);
  const int h = (b / (SEQ / 128)) % NH;
  const int n = b / ((SEQ / 128) * NH);
  const int l0 = qb * 128 + w * 16;

  const __bf16* qbase = qp + ((size_t)(n * NH + h) * SEQ + l0 + m) * HD;
  const __bf16* kbase = kp + (size_t)(n * NH + h) * SEQ * HD;
  const __bf16* vbase = vpt + (size_t)(n * NH + h) * HD * SEQ;

  v16bf aq0 = load_a_bf16(qbase, hi);
  v16bf aq1 = load_a_bf16(qbase + 32, hi);

  v8f acc[4];
  float mrow[8], lrow[8];
#pragma unroll
  for (int t = 0; t < 4; ++t)
#pragma unroll
    for (int r = 0; r < 8; ++r) acc[t][r] = 0.f;
#pragma unroll
  for (int r = 0; r < 8; ++r) { mrow[r] = -1e30f; lrow[r] = 0.f; }

  const float sc = 0.03125f;  // 1/sqrt(EMBED_DIM)

  for (int kv = 0; kv < SEQ; kv += 32) {
    if (kv + 32 < SEQ) {  // next-chunk prefetch -> global_prefetch_b8
      __builtin_prefetch(kbase + (size_t)(kv + 32 + m) * HD, 0, 1);
      __builtin_prefetch(vbase + (size_t)m * SEQ + kv + 32, 0, 1);
    }
    // S = Q * K^T  (two 16x16 column tiles, K-dim 64 in two chunks)
    v8f s0, s1;
#pragma unroll
    for (int r = 0; r < 8; ++r) { s0[r] = 0.f; s1[r] = 0.f; }
    {
      v16bf bk00 = load_b_bf16(kbase + (size_t)(kv + m) * HD + 0 + hi * 16);
      v16bf bk10 = load_b_bf16(kbase + (size_t)(kv + 16 + m) * HD + 0 + hi * 16);
      s0 = WMMA_BF16(aq0, bk00, s0);
      s1 = WMMA_BF16(aq0, bk10, s1);
      v16bf bk01 = load_b_bf16(kbase + (size_t)(kv + m) * HD + 32 + hi * 16);
      v16bf bk11 = load_b_bf16(kbase + (size_t)(kv + 16 + m) * HD + 32 + hi * 16);
      s0 = WMMA_BF16(aq1, bk01, s0);
      s1 = WMMA_BF16(aq1, bk11, s1);
    }
    // online softmax; row M lives in VGPR r across the lane half, so reduce over 16 lanes
    float p0[8], p1[8];
#pragma unroll
    for (int r = 0; r < 8; ++r) {
      float a = s0[r] * sc, bb = s1[r] * sc;
      float mx = fmaxf(a, bb);
#pragma unroll
      for (int d = 1; d < 16; d <<= 1) mx = fmaxf(mx, __shfl_xor(mx, d));
      float mn = fmaxf(mrow[r], mx);
      float scale = __expf(mrow[r] - mn);
      p0[r] = __expf(a - mn);
      p1[r] = __expf(bb - mn);
      lrow[r] = lrow[r] * scale + p0[r] + p1[r];
      mrow[r] = mn;
#pragma unroll
      for (int t = 0; t < 4; ++t) acc[t][r] *= scale;
    }
    // transpose P from C-layout to A-layout through this wave's private LDS slice
#pragma unroll
    for (int r = 0; r < 8; ++r) {
      plds[w][(r + 8 * hi) * 32 + m] = (__bf16)p0[r];
      plds[w][(r + 8 * hi) * 32 + 16 + m] = (__bf16)p1[r];
    }
    lds_fence_wave();
    v16bf ap = load_a_bf16(&plds[w][m * 32], hi);
    // acc += P(16x32) * V(32x64); V stored transposed so B loads are contiguous
#pragma unroll
    for (int t = 0; t < 4; ++t) {
      v16bf bvv = load_b_bf16(vbase + (size_t)(t * 16 + m) * SEQ + kv + hi * 16);
      acc[t] = WMMA_BF16(ap, bvv, acc[t]);
    }
  }
  // finalize: row sum of l across the 16-lane half, normalize, store bf16 [n][l][h*64+d]
  float inv[8];
#pragma unroll
  for (int r = 0; r < 8; ++r) {
    float ls = lrow[r];
#pragma unroll
    for (int d = 1; d < 16; d <<= 1) ls += __shfl_xor(ls, d);
    inv[r] = 1.f / ls;
  }
#pragma unroll
  for (int t = 0; t < 4; ++t) {
    const int e = h * HD + t * 16 + m;
#pragma unroll
    for (int r = 0; r < 8; ++r) {
      const int lr = l0 + r + 8 * hi;
      attnb[(size_t)(n * SEQ + lr) * EMB + e] = (__bf16)(acc[t][r] * inv[r]);
    }
  }
}

// ---------------- Kernel 3: output projection out = attn @ Wo^T + bo ----------------
// One wave = 16 rows x 64 output cols; K loop over EMB in chunks of 32.
__global__ void k_oproj(const __bf16* __restrict__ attnb,
                        const __bf16* __restrict__ wob,
                        const float* __restrict__ bo,
                        float* __restrict__ out) {
  const int wave = (blockIdx.x * blockDim.x + threadIdx.x) >> 5;
  const int lane = threadIdx.x & 31;
  const int m = lane & 15;
  const int hi = lane >> 4;
  const int eg = wave % (EMB / 64);
  const int rt = wave / (EMB / 64);
  const int row0 = rt * 16;
  const int e0 = eg * 64;

  v8f c[4];
#pragma unroll
  for (int t = 0; t < 4; ++t) {
    float bb = bo[e0 + t * 16 + m];
#pragma unroll
    for (int r = 0; r < 8; ++r) c[t][r] = bb;
  }
  const __bf16* arow = attnb + (size_t)(row0 + m) * EMB;
  for (int k0 = 0; k0 < EMB; k0 += 32) {
    v16bf a = load_a_bf16(arow + k0, hi);
#pragma unroll
    for (int t = 0; t < 4; ++t) {
      v16bf bw = load_b_bf16(wob + (size_t)(e0 + t * 16 + m) * EMB + k0 + hi * 16);
      c[t] = WMMA_BF16(a, bw, c[t]);
    }
  }
#pragma unroll
  for (int t = 0; t < 4; ++t)
#pragma unroll
    for (int r = 0; r < 8; ++r)
      out[(size_t)(row0 + r + 8 * hi) * EMB + e0 + t * 16 + m] = c[t][r];
}

extern "C" void kernel_launch(void* const* d_in, const int* in_sizes, int n_in,
                              void* d_out, int out_size, void* d_ws, size_t ws_size,
                              hipStream_t stream) {
  (void)in_sizes; (void)n_in; (void)out_size; (void)ws_size;
  const float* vals = (const float*)d_in[0];
  const float* keys = (const float*)d_in[1];
  const float* qry  = (const float*)d_in[2];
  const float* Wv   = (const float*)d_in[3];
  const float* bv   = (const float*)d_in[4];
  const float* Wo   = (const float*)d_in[5];
  const float* bo   = (const float*)d_in[6];
  float* out = (float*)d_out;

  // workspace layout (bf16): qp/kp/vpt 8MB each, attn 8MB, Wo-bf16 2MB = 34MB
  char* ws = (char*)d_ws;
  __bf16* qp   = (__bf16*)(ws + (size_t)0);
  __bf16* kp   = (__bf16*)(ws + ((size_t)8 << 20));
  __bf16* vpt  = (__bf16*)(ws + ((size_t)16 << 20));
  __bf16* attn = (__bf16*)(ws + ((size_t)24 << 20));
  __bf16* wob  = (__bf16*)(ws + ((size_t)32 << 20));

  k_cvt_wo<<<EMB * EMB / 256, 256, 0, stream>>>(Wo, wob);
  k_proj<<<(3 * NBATCH * NH * (SEQ / 16)) / 8, 256, 0, stream>>>(vals, keys, qry, Wv, bv, qp, kp, vpt);
  k_attn<<<NBATCH * NH * (SEQ / 128), 256, 0, stream>>>(qp, kp, vpt, attn);
  k_oproj<<<(NBATCH * (SEQ / 16) * (EMB / 64)) / 8, 256, 0, stream>>>(attn, wob, bo, out);
}
